// GeneralEdgeAttConvv1_36000415875692
// MI455X (gfx1250) — matrix operand order repaired and checked by reading, not compiled, not run on hardware
//
#include <hip/hip_runtime.h>
#include <hip/hip_bf16.h>
#include <float.h>

typedef __attribute__((ext_vector_type(16))) __bf16 v16bf;
typedef __attribute__((ext_vector_type(8)))  float  v8f;

#define IN_CH    64
#define EDGE_DIM 32
#define FEAT     96
#define OUT_CH   64
#define HEADS     4
#define NEG_SLOPE 0.2f

// ---- helpers -------------------------------------------------------------

__device__ __forceinline__ float4 ld4(const float* p) { return *(const float4*)p; }

// v16bf fragment from two spans of 8 consecutive f32 (p0 -> elems 0..7, p1 -> elems 8..15)
__device__ __forceinline__ v16bf make_frag_2x8(const float* p0, const float* p1) {
  float4 a0 = ld4(p0), a1 = ld4(p0 + 4);
  float4 b0 = ld4(p1), b1 = ld4(p1 + 4);
  v16bf v;
  v[0]  = (__bf16)a0.x; v[1]  = (__bf16)a0.y; v[2]  = (__bf16)a0.z; v[3]  = (__bf16)a0.w;
  v[4]  = (__bf16)a1.x; v[5]  = (__bf16)a1.y; v[6]  = (__bf16)a1.z; v[7]  = (__bf16)a1.w;
  v[8]  = (__bf16)b0.x; v[9]  = (__bf16)b0.y; v[10] = (__bf16)b0.z; v[11] = (__bf16)b0.w;
  v[12] = (__bf16)b1.x; v[13] = (__bf16)b1.y; v[14] = (__bf16)b1.z; v[15] = (__bf16)b1.w;
  return v;
}

// monotone float <-> uint mapping for atomic max on floats
__device__ __forceinline__ unsigned f2ord(float f) {
  unsigned u = __float_as_uint(f);
  return (u & 0x80000000u) ? ~u : (u | 0x80000000u);
}
__device__ __forceinline__ float ord2f(unsigned u) {
  return (u & 0x80000000u) ? __uint_as_float(u & 0x7fffffffu) : __uint_as_float(~u);
}

// A-fragment for K-step kk (feature = [x_row(64) | ef_row(32)]); spans never cross the boundary.
__device__ __forceinline__ v16bf make_a_frag(const float* xrow, const float* efrow, int kk, int half) {
  int s0 = kk * 32 + 16 * half;         // elems 0..7  -> K = s0..s0+7
  int s1 = s0 + 8;                      // elems 8..15 -> K = s1..s1+7
  const float* p0 = (s0 < IN_CH) ? (xrow + s0) : (efrow + (s0 - IN_CH));
  const float* p1 = (s1 < IN_CH) ? (xrow + s1) : (efrow + (s1 - IN_CH));
  return make_frag_2x8(p0, p1);
}

// Load this lane's 8 destination-node ids (rows m = r + 8*half) with wide b128 loads.
// Fast path is wave-uniform; tail tiles fall back to clamped scalar loads.
__device__ __forceinline__ void load_drow(const long long* __restrict__ ei, long long E,
                                          long long e0, int half, int (&drow)[8]) {
  if (e0 + 16 <= E) {
    const int4* dp = (const int4*)(ei + E + e0 + 8 * half);   // 16B-aligned
    int4 d01 = dp[0], d23 = dp[1], d45 = dp[2], d67 = dp[3];
    drow[0] = d01.x; drow[1] = d01.z; drow[2] = d23.x; drow[3] = d23.z;
    drow[4] = d45.x; drow[5] = d45.z; drow[6] = d67.x; drow[7] = d67.z;
  } else {
#pragma unroll
    for (int r = 0; r < 8; ++r) {
      long long ee = e0 + r + 8 * half; if (ee >= E) ee = E - 1;
      drow[r] = (int)ei[E + ee];
    }
  }
}

// ---- kernel 0: init workspace + output ----------------------------------

__global__ void k_init(unsigned* __restrict__ segmax, float* __restrict__ segsum,
                       float* __restrict__ out, long long n4, long long nout) {
  long long i = (long long)blockIdx.x * blockDim.x + threadIdx.x;
  if (i < n4) { segmax[i] = 0u; segsum[i] = 0.0f; }   // ord 0 == "below -inf"
  if (i < nout) out[i] = 0.0f;
}

// ---- kernel 1: attention logits via folded-weight WMMA + segment max ----
// alpha[m,h] = feat[m,:] . Wa[:,h],  Wa[k,h] = sum_d W[h*16+d, k] * att[h,d]

__global__ void __launch_bounds__(256)
k_alpha(const float* __restrict__ x, const long long* __restrict__ ei,
        const float* __restrict__ ef, const float* __restrict__ W,
        const float* __restrict__ att, float* __restrict__ alphaws,
        unsigned* __restrict__ segmax, long long E, int numTiles) {
  const int lane = threadIdx.x & 31;
  const int half = lane >> 4;
  const int n    = lane & 15;
  int wave   = blockIdx.x * (blockDim.x >> 5) + (threadIdx.x >> 5);
  int nWaves = gridDim.x * (blockDim.x >> 5);

  // Build folded B fragments once per wave: B[k][col] = Wa[k][col] (col<4), else 0.
  v16bf bafrag[3];
#pragma unroll
  for (int kk = 0; kk < 3; ++kk) {
    float s[16];
#pragma unroll
    for (int i = 0; i < 16; ++i) s[i] = 0.0f;
    if (n < HEADS) {
      int kbase = kk * 32 + 16 * half;           // K = kbase + i
#pragma unroll
      for (int d = 0; d < 16; ++d) {
        float av = att[n * 16 + d];
        const float* wr = W + (size_t)(n * 16 + d) * FEAT + kbase;
        float4 c[4] = { ld4(wr), ld4(wr + 4), ld4(wr + 8), ld4(wr + 12) };
#pragma unroll
        for (int q = 0; q < 4; ++q) {
          s[q * 4 + 0] += c[q].x * av; s[q * 4 + 1] += c[q].y * av;
          s[q * 4 + 2] += c[q].z * av; s[q * 4 + 3] += c[q].w * av;
        }
      }
    }
    v16bf v;
#pragma unroll
    for (int i = 0; i < 16; ++i) v[i] = (__bf16)s[i];
    bafrag[kk] = v;
  }

  for (int tile = wave; tile < numTiles; tile += nWaves) {
    long long e0 = (long long)tile * 16;
    long long em = e0 + n; if (em >= E) em = E - 1;          // clamp tail (keeps EXEC all-1)
    int srcm = (int)ei[em];
    const float* xrow  = x  + (size_t)srcm * IN_CH;
    const float* efrow = ef + (size_t)em * EDGE_DIM;

    int drow[8];
    load_drow(ei, E, e0, half, drow);

    // prefetch next tile's edge-index lines (global_prefetch_b8)
    long long eNext = (long long)(tile + nWaves) * 16;
    if (eNext + 16 <= E) {
      __builtin_prefetch(ei + eNext, 0, 0);
      __builtin_prefetch(ei + E + eNext, 0, 0);
    }

    v16bf afrag[3];
#pragma unroll
    for (int kk = 0; kk < 3; ++kk) afrag[kk] = make_a_frag(xrow, efrow, kk, half);

    v8f acc = {0.f, 0.f, 0.f, 0.f, 0.f, 0.f, 0.f, 0.f};
#pragma unroll
    for (int kk = 0; kk < 3; ++kk)
      acc = __builtin_amdgcn_wmma_f32_16x16x32_bf16(false, afrag[kk], false, bafrag[kk],
                                                    (short)0, acc, false, false);
    // lane column n == head h (h<4); row m = r + 8*half
#pragma unroll
    for (int r = 0; r < 8; ++r) {
      long long e = e0 + r + 8 * half;
      if (n < HEADS && e < E) {
        float v = acc[r];
        float a = (v >= 0.f) ? v : NEG_SLOPE * v;            // leaky relu
        alphaws[(size_t)e * HEADS + n] = a;
        atomicMax(&segmax[(size_t)drow[r] * HEADS + n], f2ord(a));
      }
    }
  }
}

// ---- kernel 2: exp(alpha - segmax) + segment sum (1 thread = 1 edge) ----

__global__ void k_softmax_sum(const long long* __restrict__ ei,
                              const unsigned* __restrict__ segmax,
                              float* __restrict__ alphaws, float* __restrict__ segsum,
                              long long E) {
  long long e = (long long)blockIdx.x * blockDim.x + threadIdx.x;
  if (e >= E) return;
  int d = (int)ei[E + e];
  float4 a = ((const float4*)alphaws)[e];
  uint4  m = ((const uint4*)segmax)[d];
  float4 r;
  r.x = expf(a.x - ord2f(m.x));
  r.y = expf(a.y - ord2f(m.y));
  r.z = expf(a.z - ord2f(m.z));
  r.w = expf(a.w - ord2f(m.w));
  ((float4*)alphaws)[e] = r;
  atomicAdd(&segsum[(size_t)d * HEADS + 0], r.x);
  atomicAdd(&segsum[(size_t)d * HEADS + 1], r.y);
  atomicAdd(&segsum[(size_t)d * HEADS + 2], r.z);
  atomicAdd(&segsum[(size_t)d * HEADS + 3], r.w);
}

// ---- kernel 3: msg GEMM (bf16 WMMA) + normalized scatter-add ------------

__global__ void __launch_bounds__(256)
k_aggregate(const float* __restrict__ x, const long long* __restrict__ ei,
            const float* __restrict__ ef, const float* __restrict__ W,
            const float* __restrict__ alphaws, const float* __restrict__ segsum,
            float* __restrict__ out, long long E, int numTiles) {
  const int lane = threadIdx.x & 31;
  const int half = lane >> 4;
  const int n    = lane & 15;
  int wave   = blockIdx.x * (blockDim.x >> 5) + (threadIdx.x >> 5);
  int nWaves = gridDim.x * (blockDim.x >> 5);

  // Hoist B fragments: B[k][col] = W[nt*16+col][kk*32+k]; lane holds col=n, K = i + 16*half
  v16bf bfrag[HEADS][3];
#pragma unroll
  for (int nt = 0; nt < HEADS; ++nt)
#pragma unroll
    for (int kk = 0; kk < 3; ++kk) {
      const float* row = W + (size_t)(nt * 16 + n) * FEAT + kk * 32 + 16 * half;
      bfrag[nt][kk] = make_frag_2x8(row, row + 8);
    }

  for (int tile = wave; tile < numTiles; tile += nWaves) {
    long long e0 = (long long)tile * 16;
    long long em = e0 + n;
    bool valid = (em < E);
    if (!valid) em = E - 1;
    int srcm = (int)ei[em];
    int dstm = (int)ei[E + em];
    const float* xrow  = x  + (size_t)srcm * IN_CH;
    const float* efrow = ef + (size_t)em * EDGE_DIM;

    int drow[8];
    load_drow(ei, E, e0, half, drow);

    // prefetch next tile's edge-index lines
    long long eNext = (long long)(tile + nWaves) * 16;
    if (eNext + 16 <= E) {
      __builtin_prefetch(ei + eNext, 0, 0);
      __builtin_prefetch(ei + E + eNext, 0, 0);
    }

    // normalized attention weight per (edge=n, head); lane n holds edge e0+n
    float4 aw = ((const float4*)alphaws)[em];
    float wnorm[HEADS];
    {
      const float* an = &aw.x;
#pragma unroll
      for (int h = 0; h < HEADS; ++h) {
        float den = segsum[(size_t)dstm * HEADS + h] + 1e-16f;
        wnorm[h] = valid ? (an[h] / den) : 0.f;
      }
    }

    v16bf afrag[3];
#pragma unroll
    for (int kk = 0; kk < 3; ++kk) afrag[kk] = make_a_frag(xrow, efrow, kk, half);

#pragma unroll
    for (int nt = 0; nt < HEADS; ++nt) {
      v8f acc = {0.f, 0.f, 0.f, 0.f, 0.f, 0.f, 0.f, 0.f};
#pragma unroll
      for (int kk = 0; kk < 3; ++kk)
        acc = __builtin_amdgcn_wmma_f32_16x16x32_bf16(false, afrag[kk], false, bfrag[nt][kk],
                                                      (short)0, acc, false, false);
#pragma unroll
      for (int r = 0; r < 8; ++r) {
        int m = r + 8 * half;                        // edge row within tile
        float w = __shfl(wnorm[nt], m, 32);          // independent shuffles, no chain
        long long e = e0 + m;
        if (e < E)
          atomicAdd(&out[(size_t)drow[r] * OUT_CH + nt * 16 + n], acc[r] * w);
      }
    }
  }
}

// ---- launch --------------------------------------------------------------

extern "C" void kernel_launch(void* const* d_in, const int* in_sizes, int n_in,
                              void* d_out, int out_size, void* d_ws, size_t ws_size,
                              hipStream_t stream) {
  const float*     x   = (const float*)d_in[0];
  const long long* ei  = (const long long*)d_in[1];
  const float*     ef  = (const float*)d_in[2];
  const float*     W   = (const float*)d_in[3];
  const float*     att = (const float*)d_in[4];
  float* out = (float*)d_out;

  long long N = (long long)in_sizes[0] / IN_CH;
  long long E = (long long)in_sizes[1] / 2;

  // carve workspace: alphaws[E*4] f32, segmax[N*4] u32, segsum[N*4] f32
  char* p = (char*)d_ws;
  float* alphaws = (float*)p;        p += ((size_t)E * HEADS * 4 + 255) & ~(size_t)255;
  unsigned* segmax = (unsigned*)p;   p += ((size_t)N * HEADS * 4 + 255) & ~(size_t)255;
  float* segsum = (float*)p;

  long long n4 = N * HEADS, nout = N * OUT_CH;
  long long initN = (n4 > nout) ? n4 : nout;
  k_init<<<(unsigned)((initN + 255) / 256), 256, 0, stream>>>(segmax, segsum, out, n4, nout);

  int numTiles = (int)((E + 15) / 16);
  int blocks = (numTiles + 7) / 8;            // 8 waves / 256-thread block
  if (blocks > 8192) blocks = 8192;           // grid-stride; hoisted fragments amortized
  k_alpha<<<blocks, 256, 0, stream>>>(x, ei, ef, W, att, alphaws, segmax, E, numTiles);

  k_softmax_sum<<<(unsigned)((E + 255) / 256), 256, 0, stream>>>(ei, segmax, alphaws, segsum, E);

  k_aggregate<<<blocks, 256, 0, stream>>>(x, ei, ef, W, alphaws, segsum, out, E, numTiles);
}
